// SwinTransformerBlock_71975061946688
// MI455X (gfx1250) — compile-verified, matrix operand order checked by reading, not run on hardware
//
#include <hip/hip_runtime.h>
#include <hip/hip_bf16.h>
#include <math.h>

typedef __bf16  bf16x16 __attribute__((ext_vector_type(16)));
typedef float   f32x8   __attribute__((ext_vector_type(8)));

#define DIM    384
#define HEADS  12
#define HD     32
#define NTOK   49
#define NPAD   64
#define HRES   56
#define NWTOT  2048            // 32 batches * 64 windows
#define M1     131072          // NWTOT * NPAD
#define TOKS   3137
#define M2     100384          // 32 * 3137
#define M2PAD  100480          // 785 * 128
#define FF     1536
#define QLEN   50331648ll      // NWTOT*HEADS*NPAD*HD

// ---------- helpers ----------
__device__ __forceinline__ unsigned short f2bf(float f) {
    unsigned u = __builtin_bit_cast(unsigned, f);
    unsigned r = (u + 0x7FFFu + ((u >> 16) & 1u)) >> 16;
    return (unsigned short)r;
}

union FragU { bf16x16 f; uint4 q[2]; };

// A/B fragment load (16x32 bf16, CDNA5 wave32 layout):
// lane: row = lane&15, K-chunks at kb=(lane>>4)*8 and 16+kb (each 8 contiguous)
__device__ __forceinline__ bf16x16 load_frag(const unsigned short* base, int stride) {
    const int lane = threadIdx.x & 31;
    const int row  = lane & 15;
    const int kb   = (lane >> 4) << 3;
    FragU u;
    u.q[0] = *reinterpret_cast<const uint4*>(base + row * stride + kb);
    u.q[1] = *reinterpret_cast<const uint4*>(base + row * stride + 16 + kb);
    return u.f;
}

__device__ __forceinline__ f32x8 wmma_bf16(bf16x16 a, bf16x16 b, f32x8 c) {
    return __builtin_amdgcn_wmma_f32_16x16x32_bf16(false, a, false, b, (short)0, c,
                                                   false, false);
}

__device__ __forceinline__ float block_sum128(float v, float* sbuf) {
    for (int o = 16; o; o >>= 1) v += __shfl_xor(v, o, 32);
    if ((threadIdx.x & 31) == 0) sbuf[threadIdx.x >> 5] = v;
    __syncthreads();
    v = sbuf[0] + sbuf[1] + sbuf[2] + sbuf[3];
    __syncthreads();
    return v;
}

// ---------- weight convert ----------
__global__ __launch_bounds__(256) void cvt_bf16_kernel(const float* __restrict__ in,
                                                       unsigned short* __restrict__ out,
                                                       long n) {
    long i = (long)blockIdx.x * 256 + threadIdx.x;
    if (i < n) out[i] = f2bf(in[i]);
}

// ---------- LN1 + shifted window partition -> bf16 A1 [M1,384] ----------
__global__ __launch_bounds__(128)
void ln1_window_kernel(const float* __restrict__ x, const float* __restrict__ g,
                       const float* __restrict__ bv, unsigned short* __restrict__ A1) {
    __shared__ float sbuf[4];
    const long row = blockIdx.x;
    const int  t   = threadIdx.x;
    unsigned short* out = A1 + row * DIM;
    const long win = row >> 6;
    const int  tok = (int)(row & 63);
    if (tok >= NTOK) { out[t] = 0; out[t + 128] = 0; out[t + 256] = 0; return; }
    const int b  = (int)(win >> 6);
    const int nw = (int)(win & 63);
    const int wy = nw >> 3, wx = nw & 7;
    const int r  = tok / 7, c = tok - r * 7;
    int oy = wy * 7 + r + 3; if (oy >= HRES) oy -= HRES;   // roll(-3): fetch (gy+3)%56
    int ox = wx * 7 + c + 3; if (ox >= HRES) ox -= HRES;
    const float* src = x + ((long)b * TOKS + 1 + (long)oy * HRES + ox) * DIM;
    float v0 = src[t], v1 = src[t + 128], v2 = src[t + 256];
    float mu = block_sum128(v0 + v1 + v2, sbuf) * (1.0f / DIM);
    float d0 = v0 - mu, d1 = v1 - mu, d2 = v2 - mu;
    float var = block_sum128(d0 * d0 + d1 * d1 + d2 * d2, sbuf) * (1.0f / DIM);
    float rs = rsqrtf(var + 1e-5f);
    out[t]       = f2bf(d0 * rs * g[t]       + bv[t]);
    out[t + 128] = f2bf(d1 * rs * g[t + 128] + bv[t + 128]);
    out[t + 256] = f2bf(d2 * rs * g[t + 256] + bv[t + 256]);
}

// ---------- LN2 : x_new -> d_out (f32) + A2 (bf16, zero-padded) ----------
__global__ __launch_bounds__(128)
void ln2_kernel(const float* __restrict__ xn, const float* __restrict__ g,
                const float* __restrict__ bv, float* __restrict__ outF,
                unsigned short* __restrict__ A2) {
    __shared__ float sbuf[4];
    const long row = blockIdx.x;
    const int  t   = threadIdx.x;
    unsigned short* a = A2 + row * DIM;
    if (row >= M2) { a[t] = 0; a[t + 128] = 0; a[t + 256] = 0; return; }
    const float* src = xn + row * DIM;
    float v0 = src[t], v1 = src[t + 128], v2 = src[t + 256];
    float mu = block_sum128(v0 + v1 + v2, sbuf) * (1.0f / DIM);
    float d0 = v0 - mu, d1 = v1 - mu, d2 = v2 - mu;
    float var = block_sum128(d0 * d0 + d1 * d1 + d2 * d2, sbuf) * (1.0f / DIM);
    float rs = rsqrtf(var + 1e-5f);
    float y0 = d0 * rs * g[t]       + bv[t];
    float y1 = d1 * rs * g[t + 128] + bv[t + 128];
    float y2 = d2 * rs * g[t + 256] + bv[t + 256];
    float* o = outF + row * DIM;
    o[t] = y0; o[t + 128] = y1; o[t + 256] = y2;
    a[t] = f2bf(y0); a[t + 128] = f2bf(y1); a[t + 256] = f2bf(y2);
}

// ---------- time token passthrough ----------
__global__ __launch_bounds__(128) void timetok_kernel(const float* __restrict__ x,
                                                      float* __restrict__ xn) {
    int i = blockIdx.x * 128 + threadIdx.x;      // 32*384
    int b = i / DIM, c = i - b * DIM;
    long idx = (long)b * TOKS * DIM + c;
    xn[idx] = x[idx];
}

// ---------- generic bf16 WMMA GEMM, 128x64 block tile, double-buffered LDS ----------
// EPI: 1=QKV scatter, 2=proj (+shortcut, un-roll scatter), 3=fc1 (GELU->bf16), 4=fc2 (+= d_out)
template <int EPI>
__global__ __launch_bounds__(256)
void gemm_bf16_kernel(const unsigned short* __restrict__ A,
                      const unsigned short* __restrict__ Bw,
                      const float* __restrict__ bias, int K, int N,
                      unsigned short* __restrict__ outU, float* __restrict__ outF,
                      const float* __restrict__ resid) {
    __shared__ unsigned short sA[2][128 * 40];
    __shared__ unsigned short sB[2][64 * 40];
    const int t    = threadIdx.x;
    const int lane = t & 31;
    const int wv   = t >> 5;
    const int wr   = wv & 3, wc = wv >> 2;
    const long blockM = (long)blockIdx.x * 128;
    const int  blockN = blockIdx.y * 64;

    // per-thread staging roles
    const int ar = t >> 1, ac = (t & 1) * 16;     // A: 128 rows x two 16-elem chunks
    const int bn = t & 63, bkg = t >> 6;          // B: 64 cols x four 8-k chunks
    const unsigned short* aSrc = A + (blockM + ar) * (long)K + ac;
    const unsigned short* bSrc = Bw + (long)bkg * 8 * N + blockN + bn;

    f32x8 acc[2][2];
    for (int a = 0; a < 2; ++a)
        for (int b = 0; b < 2; ++b)
            for (int e = 0; e < 8; ++e) acc[a][b][e] = 0.0f;

    // prologue: stage k0 = 0 tile into buffer 0
    {
        uint4 aReg = *reinterpret_cast<const uint4*>(aSrc);
        union { uint4 q; unsigned short s[8]; } bReg;
        for (int j = 0; j < 8; ++j) bReg.s[j] = bSrc[(long)j * N];
        *reinterpret_cast<uint4*>(&sA[0][ar * 40 + ac]) = aReg;
        *reinterpret_cast<uint4*>(&sB[0][bn * 40 + bkg * 8]) = bReg.q;
    }
    __syncthreads();

    int buf = 0;
    for (int k0 = 0; k0 < K; k0 += 32, buf ^= 1) {
        const int nxt = k0 + 32;
        // issue next tile's global loads before consuming current buffer
        uint4 aReg;
        union { uint4 q; unsigned short s[8]; } bReg;
        if (nxt < K) {
            aReg = *reinterpret_cast<const uint4*>(aSrc + nxt);
            for (int j = 0; j < 8; ++j) bReg.s[j] = bSrc[(long)(nxt + j) * N];
            if (nxt + 32 < K)   // pull the tile after next toward the WGP cache
                __builtin_prefetch(aSrc + nxt + 32, 0, 0);
        }

        bf16x16 a0 = load_frag(&sA[buf][(wr * 32) * 40], 40);
        bf16x16 a1 = load_frag(&sA[buf][(wr * 32 + 16) * 40], 40);
        bf16x16 b0 = load_frag(&sB[buf][(wc * 32) * 40], 40);
        bf16x16 b1 = load_frag(&sB[buf][(wc * 32 + 16) * 40], 40);
        acc[0][0] = wmma_bf16(a0, b0, acc[0][0]);
        acc[0][1] = wmma_bf16(a0, b1, acc[0][1]);
        acc[1][0] = wmma_bf16(a1, b0, acc[1][0]);
        acc[1][1] = wmma_bf16(a1, b1, acc[1][1]);

        if (nxt < K) {
            *reinterpret_cast<uint4*>(&sA[buf ^ 1][ar * 40 + ac]) = aReg;
            *reinterpret_cast<uint4*>(&sB[buf ^ 1][bn * 40 + bkg * 8]) = bReg.q;
        }
        __syncthreads();   // single barrier per K-step (double buffered)
    }

    const int colL = lane & 15;
    const int hi8  = (lane >> 4) * 8;
    for (int ti = 0; ti < 2; ++ti)
        for (int tj = 0; tj < 2; ++tj) {
            const int col = blockN + wc * 32 + tj * 16 + colL;
            const float bvv = bias[col];
            for (int i = 0; i < 8; ++i) {
                const long row = blockM + wr * 32 + ti * 16 + hi8 + i;
                float val = acc[ti][tj][i] + bvv;
                if constexpr (EPI == 1) {            // QKV scatter
                    int part = col / DIM;
                    int rem  = col - part * DIM;
                    int head = rem >> 5, d = rem & 31;
                    long win = row >> 6; int tok = (int)(row & 63);
                    if (part == 0) val *= 0.17677669529663687f;  // 1/sqrt(32)
                    unsigned short* dst = outU + (long)part * QLEN;
                    dst[((win * HEADS + head) * NPAD + tok) * HD + d] = f2bf(val);
                } else if constexpr (EPI == 2) {     // proj: un-window + roll + shortcut
                    long win = row >> 6; int tok = (int)(row & 63);
                    if (tok < NTOK) {
                        int b  = (int)(win >> 6);
                        int nw = (int)(win & 63);
                        int wy = nw >> 3, wx = nw & 7;
                        int r = tok / 7, c = tok - r * 7;
                        int oy = wy * 7 + r + 3; if (oy >= HRES) oy -= HRES;
                        int ox = wx * 7 + c + 3; if (ox >= HRES) ox -= HRES;
                        long idx = ((long)b * TOKS + 1 + (long)oy * HRES + ox) * DIM + col;
                        outF[idx] = resid[idx] + val;
                    }
                } else if constexpr (EPI == 3) {     // fc1: exact GELU -> bf16
                    float gl = 0.5f * val * (1.0f + erff(val * 0.70710678118654752f));
                    outU[row * (long)N + col] = f2bf(gl);
                } else {                              // fc2: accumulate into d_out
                    if (row < M2) outF[row * (long)DIM + col] += val;
                }
            }
        }
}

// ---------- attention: one block per (window, head) ----------
__global__ __launch_bounds__(128)
void attn_kernel(const unsigned short* __restrict__ q, const unsigned short* __restrict__ k,
                 const unsigned short* __restrict__ v, const float* __restrict__ rpb,
                 unsigned short* __restrict__ attn_out) {
    __shared__ unsigned short sQ[64 * 40];
    __shared__ unsigned short sK[64 * 40];
    __shared__ unsigned short sVt[32 * 72];
    __shared__ unsigned short sP[64 * 72];

    const int wh   = blockIdx.x;           // win*HEADS + head
    const int head = wh % HEADS;
    const int win  = wh / HEADS;
    const int nw   = win & 63;
    const int wy   = nw >> 3, wx = nw & 7;
    const int t    = threadIdx.x;
    const int lane = t & 31;
    const int wv   = t >> 5;

    { // stage q,k (row-major 64x32) and v transposed (32x64)
        int rowq = t >> 1, off = (t & 1) * 16;
        *reinterpret_cast<uint4*>(&sQ[rowq * 40 + off]) =
            *reinterpret_cast<const uint4*>(q + ((long)wh * NPAD + rowq) * HD + off);
        *reinterpret_cast<uint4*>(&sK[rowq * 40 + off]) =
            *reinterpret_cast<const uint4*>(k + ((long)wh * NPAD + rowq) * HD + off);
        int d = t >> 2, tk0 = (t & 3) * 16;
        for (int j = 0; j < 16; ++j)
            sVt[d * 72 + tk0 + j] = v[((long)wh * NPAD + tk0 + j) * HD + d];
    }
    __syncthreads();

    const int rb0  = 16 * wv;              // this wave's 16 query rows
    const int colL = lane & 15;
    const int hi8  = (lane >> 4) * 8;

    // S = q @ k^T   (K=32 = head_dim, one WMMA step; 4 column tiles)
    bf16x16 qf = load_frag(&sQ[rb0 * 40], 40);
    f32x8 accS[4];
    for (int t4 = 0; t4 < 4; ++t4) {
        for (int e = 0; e < 8; ++e) accS[t4][e] = 0.0f;
        bf16x16 kf = load_frag(&sK[(t4 * 16) * 40], 40);
        accS[t4] = wmma_bf16(qf, kf, accS[t4]);
    }

    // bias + shift-mask + softmax (rows spread over 16 lanes, 4 regs)
    for (int i = 0; i < 8; ++i) {
        const int n_tok = rb0 + hi8 + i;
        int nr = 0, nc = 0, regn = 0;
        if (n_tok < NTOK) {
            nr = n_tok / 7; nc = n_tok - nr * 7;
            int gy = wy * 7 + nr, gx = wx * 7 + nc;  // mask uses shifted-frame coords
            regn = ((gy < 49) ? 0 : ((gy < 53) ? 1 : 2)) * 3 +
                   ((gx < 49) ? 0 : ((gx < 53) ? 1 : 2));
        }
        float sv[4];
        for (int t4 = 0; t4 < 4; ++t4) {
            const int m_tok = t4 * 16 + colL;
            float s = accS[t4][i];
            if (n_tok < NTOK && m_tok < NTOK) {
                int mr = m_tok / 7, mc = m_tok - mr * 7;
                s += rpb[((nr - mr + 6) * 13 + (nc - mc + 6)) * HEADS + head];
                int gy = wy * 7 + mr, gx = wx * 7 + mc;
                int regm = ((gy < 49) ? 0 : ((gy < 53) ? 1 : 2)) * 3 +
                           ((gx < 49) ? 0 : ((gx < 53) ? 1 : 2));
                if (regm != regn) s -= 100.0f;
            } else {
                s = -1e30f;                 // padded cols -> exp() == 0
            }
            sv[t4] = s;
        }
        float mx = fmaxf(fmaxf(sv[0], sv[1]), fmaxf(sv[2], sv[3]));
        for (int o = 1; o < 16; o <<= 1) mx = fmaxf(mx, __shfl_xor(mx, o, 32));
        float ev[4], sum = 0.0f;
        for (int t4 = 0; t4 < 4; ++t4) { ev[t4] = expf(sv[t4] - mx); sum += ev[t4]; }
        for (int o = 1; o < 16; o <<= 1) sum += __shfl_xor(sum, o, 32);
        float inv = 1.0f / sum;
        const int prow = (rb0 + hi8 + i) * 72;
        for (int t4 = 0; t4 < 4; ++t4)
            sP[prow + t4 * 16 + colL] = f2bf(ev[t4] * inv);
    }
    // wave-private rows: no __syncthreads needed before re-reading sP

    // O = P @ V   (K = 64 tokens, 2 steps; N = 32 = head_dim, 2 tiles)
    f32x8 accO[2];
    for (int tj = 0; tj < 2; ++tj)
        for (int e = 0; e < 8; ++e) accO[tj][e] = 0.0f;
    for (int ks = 0; ks < 2; ++ks) {
        bf16x16 pf = load_frag(&sP[rb0 * 72 + ks * 32], 72);
        for (int tj = 0; tj < 2; ++tj) {
            bf16x16 vf = load_frag(&sVt[(tj * 16) * 72 + ks * 32], 72);
            accO[tj] = wmma_bf16(pf, vf, accO[tj]);
        }
    }
    for (int tj = 0; tj < 2; ++tj)
        for (int i = 0; i < 8; ++i) {
            const int tok = rb0 + hi8 + i;
            attn_out[((long)win * NPAD + tok) * DIM + head * HD + tj * 16 + colL] =
                f2bf(accO[tj][i]);
        }
}

// ---------- host ----------
extern "C" void kernel_launch(void* const* d_in, const int* in_sizes, int n_in,
                              void* d_out, int out_size, void* d_ws, size_t ws_size,
                              hipStream_t stream) {
    (void)in_sizes; (void)n_in; (void)out_size; (void)ws_size;
    const float* x      = (const float*)d_in[0];
    const float* n1g    = (const float*)d_in[1];
    const float* n1b    = (const float*)d_in[2];
    const float* qkv_w  = (const float*)d_in[3];
    const float* qkv_b  = (const float*)d_in[4];
    const float* rpb    = (const float*)d_in[5];
    const float* proj_w = (const float*)d_in[6];
    const float* proj_b = (const float*)d_in[7];
    const float* n2g    = (const float*)d_in[8];
    const float* n2b    = (const float*)d_in[9];
    const float* fc1_w  = (const float*)d_in[10];
    const float* fc1_b  = (const float*)d_in[11];
    const float* fc2_w  = (const float*)d_in[12];
    const float* fc2_b  = (const float*)d_in[13];
    float* out = (float*)d_out;

    char* p = (char*)d_ws;
    auto take = [&p](size_t bytes) {
        char* r = p; p += (bytes + 255) & ~(size_t)255; return r;
    };
    unsigned short* wqkv = (unsigned short*)take((size_t)DIM * 3 * DIM * 2);
    unsigned short* wprj = (unsigned short*)take((size_t)DIM * DIM * 2);
    unsigned short* wf1  = (unsigned short*)take((size_t)DIM * FF * 2);
    unsigned short* wf2  = (unsigned short*)take((size_t)FF * DIM * 2);
    char* rbase = p;                                       // H aliases this region
    unsigned short* A1   = (unsigned short*)take((size_t)M1 * DIM * 2);   // also attn_out
    unsigned short* qb   = (unsigned short*)take((size_t)3 * QLEN * 2);
    float*          xnew = (float*)take((size_t)M2 * DIM * 4);
    unsigned short* H    = (unsigned short*)rbase;         // M2PAD*FF bf16 fits in region
    unsigned short* A2   = (unsigned short*)take((size_t)M2PAD * DIM * 2);
    unsigned short* kb   = qb + QLEN;
    unsigned short* vb   = kb + QLEN;

    auto cvt = [&](const float* src, unsigned short* dst, long n) {
        cvt_bf16_kernel<<<(unsigned)((n + 255) / 256), 256, 0, stream>>>(src, dst, n);
    };
    cvt(qkv_w, wqkv, (long)DIM * 3 * DIM);
    cvt(proj_w, wprj, (long)DIM * DIM);
    cvt(fc1_w, wf1, (long)DIM * FF);
    cvt(fc2_w, wf2, (long)FF * DIM);

    ln1_window_kernel<<<M1, 128, 0, stream>>>(x, n1g, n1b, A1);

    gemm_bf16_kernel<1><<<dim3(M1 / 128, (3 * DIM) / 64), 256, 0, stream>>>(
        A1, wqkv, qkv_b, DIM, 3 * DIM, qb, nullptr, nullptr);

    attn_kernel<<<NWTOT * HEADS, 128, 0, stream>>>(qb, kb, vb, rpb, A1 /*attn_out*/);

    timetok_kernel<<<(32 * DIM) / 128, 128, 0, stream>>>(x, xnew);

    gemm_bf16_kernel<2><<<dim3(M1 / 128, DIM / 64), 256, 0, stream>>>(
        A1, wprj, proj_b, DIM, DIM, nullptr, xnew, x);

    ln2_kernel<<<M2PAD, 128, 0, stream>>>(xnew, n2g, n2b, out, A2);

    gemm_bf16_kernel<3><<<dim3(M2PAD / 128, FF / 64), 256, 0, stream>>>(
        A2, wf1, fc1_b, DIM, FF, H, nullptr, nullptr);

    gemm_bf16_kernel<4><<<dim3(M2PAD / 128, DIM / 64), 256, 0, stream>>>(
        H, wf2, fc2_b, FF, DIM, nullptr, out, nullptr);
}
